// DBM_71064528879811
// MI455X (gfx1250) — matrix operand order repaired and checked by reading, not compiled, not run on hardware
//
#include <hip/hip_runtime.h>

#define NN 6144          // NV == NH
#define NB 32            // batch
#define NE (NB * NN)
#define KSPLIT 4

typedef __bf16 bf16;
typedef bf16  v16bf __attribute__((ext_vector_type(16)));
typedef bf16  v8bf  __attribute__((ext_vector_type(8)));
typedef float v8f   __attribute__((ext_vector_type(8)));

__device__ __forceinline__ bf16 f2bf(float f) {
  union { float f; unsigned u; } c; c.f = f;
  unsigned r = (c.u + 0x7fffu + ((c.u >> 16) & 1u)) >> 16;  // round-nearest-even
  unsigned short s = (unsigned short)r;
  bf16 b; __builtin_memcpy(&b, &s, 2); return b;
}

__device__ __forceinline__ float rand01(unsigned x) {
  x ^= x >> 17; x *= 0xed5ad4bbu;
  x ^= x >> 11; x *= 0xac4c1b51u;
  x ^= x >> 15; x *= 0x31848babu;
  x ^= x >> 14;
  return (float)(x >> 8) * (1.0f / 16777216.0f);
}

// ---------------------------------------------------------------------------
// One-time: fp32 W (NN x NN) -> bf16 W and bf16 W^T (LDS tile transpose).
// Each orientation is 75.5 MB; the B operand of every GEMM stays L2-resident
// (192 MB L2) instead of streaming 151 MB of fp32 from HBM per GEMM.
// ---------------------------------------------------------------------------
__global__ __launch_bounds__(256) void cvt_transpose_kernel(
    const float* __restrict__ W, bf16* __restrict__ Wbf, bf16* __restrict__ WT) {
  __shared__ bf16 tile[32][33];
  const int tx = threadIdx.x & 31;
  const int ty = threadIdx.x >> 5;              // 0..7
  const int bx = blockIdx.x, by = blockIdx.y;   // 192 x 192 tiles
  const int col = bx * 32 + tx;
#pragma unroll
  for (int r = 0; r < 32; r += 8) {
    const int row = by * 32 + ty + r;
    bf16 b = f2bf(W[(size_t)row * NN + col]);
    Wbf[(size_t)row * NN + col] = b;
    tile[ty + r][tx] = b;
  }
  __syncthreads();
#pragma unroll
  for (int r = 0; r < 32; r += 8) {
    const int orow = bx * 32 + ty + r;          // original column
    const int ocol = by * 32 + tx;              // original row
    WT[(size_t)orow * NN + ocol] = tile[tx][ty + r];
  }
}

// ---------------------------------------------------------------------------
// Fused dual GEMM: C(32xNN,f32) = A1@B1 (+ A2@B2) + bias.
// A* are bf16 32xNN row-major activations (values exactly 0/1).
// B* are bf16 column-major (Bc[n*NN + k]) so each lane's B-fragment is 16
// contiguous bf16 (32B). A-fragment uses the same k = 16*hi + [0..15] order = two contiguous 16B
// chunks per lane. 4 waves split K; LDS reduction combines partials.
// Branch-free epilogue (bias always present, no C read-modify-write).
// ---------------------------------------------------------------------------
__device__ __forceinline__ void wmma_kloop(
    const bf16* __restrict__ arow, const bf16* __restrict__ bcol,
    int k0, int k1, int hi, v8f& acc) {
  for (int kk = k0; kk < k1; kk += 32) {
    __builtin_prefetch(bcol + kk + 2048, 0, 1);
    union { v16bf v; v8bf h[2]; } a;
    a.h[0] = *(const v8bf*)(arow + kk + hi * 16);        // K = hi*16 + [0..7]
    a.h[1] = *(const v8bf*)(arow + kk + hi * 16 + 8);    // K = hi*16 + 8 + [0..7]
    v16bf bv = *(const v16bf*)(bcol + kk);               // K = hi*16 + [0..15]
    acc = __builtin_amdgcn_wmma_f32_16x16x32_bf16(
        false, a.v, false, bv, (short)0, acc, false, false);
  }
}

__global__ __launch_bounds__(128) void gemm_wmma_kernel(
    const bf16* __restrict__ A1, const bf16* __restrict__ B1,
    const bf16* __restrict__ A2, const bf16* __restrict__ B2,
    const float* __restrict__ bias, float* __restrict__ C) {
  const int lane = threadIdx.x & 31;
  const int wave = threadIdx.x >> 5;            // 0..3 : K slice
  const int lo = lane & 15, hi = lane >> 4;
  const int ntile = blockIdx.x;                 // 0..NN/16-1
  const int mtile = blockIdx.y;                 // 0..1
  const int m = mtile * 16 + lo;
  const int n = ntile * 16 + lo;
  const int k0 = wave * (NN / KSPLIT);
  const int k1 = k0 + (NN / KSPLIT);

  v8f acc = {};
  wmma_kloop(A1 + (size_t)m * NN, B1 + (size_t)n * NN + hi * 16, k0, k1, hi, acc);
  if (A2)
    wmma_kloop(A2 + (size_t)m * NN, B2 + (size_t)n * NN + hi * 16, k0, k1, hi, acc);

  __shared__ float red[KSPLIT][8][32];          // [wave][r][lane]: conflict-free
#pragma unroll
  for (int r = 0; r < 8; ++r) red[wave][r][lane] = acc[r];
  __syncthreads();
  if (wave == 0) {
    const float bn = bias[n];
#pragma unroll
    for (int r = 0; r < 8; ++r) {
      float s = red[0][r][lane] + red[1][r][lane] + red[2][r][lane] + red[3][r][lane];
      const int row = mtile * 16 + hi * 8 + r;  // C layout: vgpr r -> M = 8*hi + r
      C[(size_t)row * NN + n] = s + bn;
    }
  }
}

// ---------------------------------------------------------------------------
// Elementwise / reduction helpers (activation state is bf16 0/1)
// ---------------------------------------------------------------------------
__global__ void f2b_kernel(const float* __restrict__ vimg, bf16* __restrict__ vout, int n) {
  int idx = blockIdx.x * 256 + threadIdx.x;
  if (idx >= n) return;
  int m = idx / NN, j = idx - m * NN;
  int pix = j >> 3, bit = j & 7;
  float x = vimg[m * (NN / 8) + pix];
  float qf = floorf(x * 256.0f);
  int q = (int)fminf(fmaxf(qf, 0.0f), 255.0f);
  vout[idx] = f2bf(((q >> bit) & 1) ? 1.0f : 0.0f);
}

__global__ void randbin_kernel(bf16* __restrict__ o, int n, unsigned seed) {
  int idx = blockIdx.x * 256 + threadIdx.x;
  if (idx >= n) return;
  o[idx] = f2bf((rand01((unsigned)idx * 2654435761u + seed) < 0.5f) ? 1.0f : 0.0f);
}

__global__ void samp_kernel(const float* __restrict__ lg, bf16* __restrict__ o,
                            int n, unsigned seed, int stochastic) {
  int idx = blockIdx.x * 256 + threadIdx.x;
  if (idx >= n) return;
  float l = lg[idx];
  int r;
  if (stochastic) {
    float pp = 1.0f / (1.0f + __expf(-l));
    r = rand01((unsigned)idx * 0x9E3779B9u + seed) < pp;
  } else {
    r = l >= 0.0f;
  }
  o[idx] = f2bf(r ? 1.0f : 0.0f);
}

__global__ void even_kernel(float* __restrict__ evenm, unsigned seed) {
  int m = threadIdx.x;
  if (m < NB) evenm[m] = (rand01((unsigned)m * 0x85ebca6bu + seed) < 0.5f) ? 1.0f : 0.0f;
}

__global__ void select_kernel(const float* __restrict__ evenm, const bf16* __restrict__ xe,
                              const bf16* __restrict__ xo, bf16* __restrict__ o, int n) {
  int idx = blockIdx.x * 256 + threadIdx.x;
  if (idx >= n) return;
  int m = idx / NN;
  o[idx] = (evenm[m] > 0.5f) ? xe[idx] : xo[idx];
}

// e[m] = -( v.b0 + h0.(v@W0^T+b1) + h1.(h0@W1^T+b2) ); t0/t1 hold the logits
__global__ __launch_bounds__(256) void energy_kernel(
    const bf16* __restrict__ v, const bf16* __restrict__ h0, const bf16* __restrict__ h1,
    const float* __restrict__ b0, const float* __restrict__ t0, const float* __restrict__ t1,
    float* __restrict__ e) {
  const int m = blockIdx.x;
  const int tid = threadIdx.x;
  float s = 0.0f;
  for (int j = tid; j < NN; j += 256) {
    const size_t i = (size_t)m * NN + j;
    s += (float)v[i] * b0[j] + (float)h0[i] * t0[i] + (float)h1[i] * t1[i];
  }
  __shared__ float red[256];
  red[tid] = s;
  __syncthreads();
  for (int o = 128; o > 0; o >>= 1) {
    if (tid < o) red[tid] += red[tid + o];
    __syncthreads();
  }
  if (tid == 0) e[m] = -red[0];
}

__global__ void accept_kernel(float* __restrict__ e_state, const float* __restrict__ e_prop,
                              float* __restrict__ e_accum, float* __restrict__ conv,
                              float* __restrict__ take, unsigned seed, int first) {
  int m = threadIdx.x;
  if (m >= NB) return;
  float lr = e_state[m] - e_prop[m];                 // log_ratio = E(cur) - E(prop)
  float r  = rand01((unsigned)m * 0xC2B2AE35u + seed);
  int a = r < __expf(fminf(lr, 0.0f));
  int convd = conv[m] > 0.5f;
  int tk;
  if (first) {
    tk = a;
    if (a) e_state[m] = e_prop[m];
    e_accum[m] = e_state[m];                         // e = energy after first MH step
    conv[m] = a ? 0.0f : 1.0f;                       // unchanged state => converged
  } else {
    tk = a && !convd;
    if (!convd) {
      if (a) { e_accum[m] += e_prop[m] - e_state[m]; e_state[m] = e_prop[m]; }
      else   { conv[m] = 1.0f; }
    }
  }
  take[m] = tk ? 1.0f : 0.0f;
}

__global__ void apply_kernel(bf16* __restrict__ x, const bf16* __restrict__ xp,
                             const float* __restrict__ take, int n) {
  int idx = blockIdx.x * 256 + threadIdx.x;
  if (idx >= n) return;
  int m = idx / NN;
  if (take[m] > 0.5f) x[idx] = xp[idx];
}

__global__ void diff_kernel(const float* __restrict__ ep, const float* __restrict__ en,
                            float* __restrict__ o) {
  int m = threadIdx.x;
  if (m < NB) o[m] = ep[m] - en[m];
}

// ---------------------------------------------------------------------------
// Host orchestration (all launches on `stream`; graph-capture safe)
// ---------------------------------------------------------------------------
extern "C" void kernel_launch(void* const* d_in, const int* in_sizes, int n_in,
                              void* d_out, int out_size, void* d_ws, size_t ws_size,
                              hipStream_t stream) {
  (void)in_sizes; (void)n_in; (void)out_size; (void)ws_size;
  const float* vimg = (const float*)d_in[0];
  const float* W0   = (const float*)d_in[1];
  const float* W1   = (const float*)d_in[2];
  const float* b0   = (const float*)d_in[3];
  const float* b1   = (const float*)d_in[4];
  const float* b2   = (const float*)d_in[5];
  float* out = (float*)d_out;

  // -------- workspace layout --------
  char* p = (char*)d_ws;
  auto alloc = [&](size_t bytes) { char* r = p; p += (bytes + 255) & ~(size_t)255; return r; };
  const size_t WB = (size_t)NN * NN * sizeof(bf16);
  bf16* W0bf = (bf16*)alloc(WB);   // W0 row-major bf16  -> col-major B for x@W0^T
  bf16* W0T  = (bf16*)alloc(WB);   // W0^T row-major     -> col-major B for x@W0
  bf16* W1bf = (bf16*)alloc(WB);
  bf16* W1T  = (bf16*)alloc(WB);
  const size_t SB = (size_t)NE * sizeof(bf16);   // activation states in bf16
  bf16* v   = (bf16*)alloc(SB);
  bf16* h0  = (bf16*)alloc(SB);
  bf16* h1  = (bf16*)alloc(SB);
  bf16* ve  = (bf16*)alloc(SB);
  bf16* vo  = (bf16*)alloc(SB);
  bf16* h0e = (bf16*)alloc(SB);
  bf16* h0o = (bf16*)alloc(SB);
  bf16* h1e = (bf16*)alloc(SB);
  bf16* h1o = (bf16*)alloc(SB);
  bf16* vp  = (bf16*)alloc(SB);
  bf16* h0p = (bf16*)alloc(SB);
  bf16* h1p = (bf16*)alloc(SB);
  float* t0 = (float*)alloc((size_t)NE * 4);     // logits stay fp32
  float* t1 = (float*)alloc((size_t)NE * 4);
  float* evenm   = (float*)alloc(NB * 4);
  float* e_state = (float*)alloc(NB * 4);
  float* e_prop  = (float*)alloc(NB * 4);
  float* e_pos   = (float*)alloc(NB * 4);
  float* e_neg   = (float*)alloc(NB * 4);
  float* convf   = (float*)alloc(NB * 4);
  float* takef   = (float*)alloc(NB * 4);

  unsigned seed = 0x12345678u;
  auto nextseed = [&]() { seed = seed * 747796405u + 2891336453u; return seed; };

  // -------- one-time weight conversion: fp32 -> bf16 + transpose --------
  dim3 gW(NN / 32, NN / 32);
  cvt_transpose_kernel<<<gW, 256, 0, stream>>>(W0, W0bf, W0T);
  cvt_transpose_kernel<<<gW, 256, 0, stream>>>(W1, W1bf, W1T);

  const int EG = (NE + 255) / 256;
  auto gemm = [&](const bf16* A1, const bf16* B1, const bf16* A2, const bf16* B2,
                  const float* bias, float* C) {
    gemm_wmma_kernel<<<dim3(NN / 16, 2), 32 * KSPLIT, 0, stream>>>(A1, B1, A2, B2, bias, C);
  };
  auto samp = [&](const float* lg, bf16* o, int stoch) {
    samp_kernel<<<EG, 256, 0, stream>>>(lg, o, NE, nextseed(), stoch);
  };
  auto randbin = [&](bf16* o) {
    randbin_kernel<<<EG, 256, 0, stream>>>(o, NE, nextseed());
  };
  auto sel = [&](const bf16* xe, const bf16* xo, bf16* o) {
    select_kernel<<<EG, 256, 0, stream>>>(evenm, xe, xo, o, NE);
  };
  auto energy = [&](const bf16* xv, const bf16* xh0, const bf16* xh1, float* e) {
    gemm(xv,  W0bf, nullptr, nullptr, b1, t0);       // v @ W0^T + b1
    gemm(xh0, W1bf, nullptr, nullptr, b2, t1);       // h0 @ W1^T + b2
    energy_kernel<<<NB, 256, 0, stream>>>(xv, xh0, xh1, b0, t0, t1, e);
  };
  auto gibbs = [&](int fix_v, int stoch) {
    const bf16* v_e = v;
    if (!fix_v) { gemm(h0, W0T, nullptr, nullptr, b0, t0); samp(t0, ve, stoch); v_e = ve; }
    gemm(h0, W1bf, nullptr, nullptr, b2, t1); samp(t1, h1e, stoch);        // h1_e
    gemm(v_e, W0bf, h1e, W1T, b1, t0);        samp(t0, h0e, stoch);        // h0_e (fused)
    gemm(v,   W0bf, h1,  W1T, b1, t0);        samp(t0, h0o, stoch);        // h0_o (fused)
    if (!fix_v) { gemm(h0o, W0T, nullptr, nullptr, b0, t0); samp(t0, vo, stoch); }
    gemm(h0o, W1bf, nullptr, nullptr, b2, t1); samp(t1, h1o, stoch);       // h1_o
    if (!fix_v) sel(ve, vo, v);
    sel(h0e, h0o, h0);
    sel(h1e, h1o, h1);
  };
  auto run_phase = [&](int fix_v, float* e_out) {
    if (fix_v) f2b_kernel<<<EG, 256, 0, stream>>>(vimg, v, NE);
    else       randbin(v);
    randbin(h0); randbin(h1);
    // local search: K_LS deterministic Gibbs steps (shared even mask)
    even_kernel<<<1, 32, 0, stream>>>(evenm, nextseed());
    for (int i = 0; i < 32; ++i) gibbs(fix_v, 0);
    // one stochastic Gibbs step (T=1)
    even_kernel<<<1, 32, 0, stream>>>(evenm, nextseed());
    gibbs(fix_v, 1);
    // coupling: 1 + K_CP Metropolis-Hastings steps (energies tracked incrementally)
    energy(v, h0, h1, e_state);
    for (int i = 0; i < 17; ++i) {
      const bf16* pv = v;
      if (!fix_v) { randbin(vp); pv = vp; }
      randbin(h0p); randbin(h1p);
      energy(pv, h0p, h1p, e_prop);
      accept_kernel<<<1, 32, 0, stream>>>(e_state, e_prop, e_out, convf, takef,
                                          nextseed(), i == 0);
      if (!fix_v) apply_kernel<<<EG, 256, 0, stream>>>(v, vp, takef, NE);
      apply_kernel<<<EG, 256, 0, stream>>>(h0, h0p, takef, NE);
      apply_kernel<<<EG, 256, 0, stream>>>(h1, h1p, takef, NE);
    }
  };

  run_phase(1, e_pos);   // positive phase (fix_v)
  run_phase(0, e_neg);   // negative phase
  diff_kernel<<<1, 32, 0, stream>>>(e_pos, e_neg, out);
}
